// PredictYOLO_v3_37460704755799
// MI455X (gfx1250) — compile-verified, matrix operand order checked by reading, not run on hardware
//
#include <hip/hip_runtime.h>

// ---------------- problem constants ----------------
#define BDIM    32
#define NANCH   22743
#define NCLS    80
#define TOPKN   1000
#define KSLOTS  1024
#define CONF_THR 0.05f
#define NMS_THR  0.5f
#define BOX_OFF  10.0f

typedef __attribute__((ext_vector_type(2)))  float    v2f;
typedef __attribute__((ext_vector_type(8)))  float    v8f;
typedef __attribute__((ext_vector_type(16))) _Float16 v16h;

struct SelState {
  unsigned b0;       // pass0 selected 11-bit prefix
  unsigned pref21;   // pass1: top 22 bits prefix
  unsigned kRem;     // remaining k after each pass
  unsigned thrKey;   // exact bit-key of the 1000th score
  unsigned tieNeed;  // #ties at thrKey to include
  unsigned G;        // #elements strictly greater than thrKey
  unsigned cntHigh;
  unsigned cntTie;
};

__device__ __forceinline__ float sigmoidf_(float x) {
  return 1.0f / (1.0f + __expf(-x));
}

// ---------------- 0: zero histogram + state ----------------
__global__ void k_zero(unsigned* __restrict__ hist, SelState* __restrict__ st) {
  int t = blockIdx.x * blockDim.x + threadIdx.x;
  if (t < 2048) hist[t] = 0u;
  if (t == 0) {
    st->b0 = 0u; st->pref21 = 0u; st->kRem = TOPKN; st->thrKey = 0u;
    st->tieNeed = 0u; st->G = 0u; st->cntHigh = 0u; st->cntTie = 0u;
  }
}

// ---------------- 1: objectness scores ----------------
__global__ void k_scores(const float* __restrict__ p, float* __restrict__ scores,
                         long long n) {
  long long i = (long long)blockIdx.x * blockDim.x + threadIdx.x;
  if (i >= n) return;
  float c = sigmoidf_(p[i * 85 + 4]);
  scores[i] = (c > CONF_THR) ? c : 0.0f;
}

// ---------------- 2: radix-select histogram (LDS privatized) ----------------
__global__ void k_hist(const float* __restrict__ scores, unsigned* __restrict__ hist,
                       const SelState* __restrict__ st, int pass, long long n) {
  __shared__ unsigned lh[2048];
  for (int t = threadIdx.x; t < 2048; t += blockDim.x) lh[t] = 0u;
  __syncthreads();
  unsigned b0 = st->b0;
  unsigned pref21 = st->pref21;
  long long stride = (long long)gridDim.x * blockDim.x;
  for (long long i = (long long)blockIdx.x * blockDim.x + threadIdx.x; i < n; i += stride) {
    unsigned key = __float_as_uint(scores[i]);   // scores >= 0 -> monotonic bits
    int bin = -1;
    if (pass == 0)      bin = (int)(key >> 21);
    else if (pass == 1) { if ((key >> 21) == b0)     bin = (int)((key >> 10) & 0x7FFu); }
    else                { if ((key >> 10) == pref21) bin = (int)(key & 0x3FFu); }
    if (bin >= 0) atomicAdd(&lh[bin], 1u);
  }
  __syncthreads();
  for (int t = threadIdx.x; t < 2048; t += blockDim.x)
    if (lh[t]) atomicAdd(&hist[t], lh[t]);
}

// ---------------- 3: scan histogram from top, refine prefix ----------------
__global__ void k_scan(unsigned* __restrict__ hist, SelState* __restrict__ st, int pass) {
  if (threadIdx.x != 0 || blockIdx.x != 0) return;
  int nb = (pass == 2) ? 1024 : 2048;
  unsigned k = (pass == 0) ? (unsigned)TOPKN : st->kRem;
  unsigned cnt = 0u; int sel = 0; bool found = false;
  for (int b = nb - 1; b >= 0; --b) {
    unsigned h = hist[b];
    if (cnt + h >= k) { sel = b; found = true; break; }
    cnt += h;
  }
  unsigned kNew = found ? (k - cnt) : 0u;
  if (pass == 0)      { st->b0 = (unsigned)sel; st->kRem = kNew; }
  else if (pass == 1) { st->pref21 = (st->b0 << 11) | (unsigned)sel; st->kRem = kNew; }
  else {
    st->thrKey  = (st->pref21 << 10) | (unsigned)sel;
    st->tieNeed = kNew;
    st->G       = (unsigned)TOPKN - kNew;
    st->cntHigh = 0u; st->cntTie = 0u;
  }
  for (int b = 0; b < 2048; ++b) hist[b] = 0u;   // ready for next pass
}

// ---------------- 4: init candidate slots ----------------
__global__ void k_initcand(float* __restrict__ candS, int* __restrict__ candI) {
  int t = threadIdx.x;
  candS[t] = -1.0f;
  candI[t] = -1;
}

// ---------------- 5: compaction into <=1024 candidates ----------------
__global__ void k_compact(const float* __restrict__ scores, float* __restrict__ candS,
                          int* __restrict__ candI, SelState* __restrict__ st, long long n) {
  unsigned thr = st->thrKey;
  unsigned G = st->G, tieNeed = st->tieNeed;
  long long stride = (long long)gridDim.x * blockDim.x;
  for (long long i = (long long)blockIdx.x * blockDim.x + threadIdx.x; i < n; i += stride) {
    float s = scores[i];
    unsigned key = __float_as_uint(s);
    if (key > thr) {
      unsigned p = atomicAdd(&st->cntHigh, 1u);
      if (p < (unsigned)KSLOTS) { candS[p] = s; candI[p] = (int)i; }
    } else if (key == thr) {
      unsigned t = atomicAdd(&st->cntTie, 1u);
      if (t < tieNeed) {
        unsigned p = G + t;
        if (p < (unsigned)KSLOTS) { candS[p] = s; candI[p] = (int)i; }
      }
    }
  }
}

// ---------------- 6: bitonic sort (desc by score, asc idx) ----------------
__global__ void __launch_bounds__(KSLOTS) k_sort(float* __restrict__ candS,
                                                 int* __restrict__ candI) {
  __shared__ float ss[KSLOTS];
  __shared__ int   si[KSLOTS];
  int tid = threadIdx.x;
  ss[tid] = candS[tid]; si[tid] = candI[tid];
  __syncthreads();
  for (int k = 2; k <= KSLOTS; k <<= 1) {
    for (int j = k >> 1; j > 0; j >>= 1) {
      int ixj = tid ^ j;
      if (ixj > tid) {
        float s1 = ss[tid], s2 = ss[ixj];
        int   i1 = si[tid], i2 = si[ixj];
        bool firstBefore = (s1 > s2) || (s1 == s2 && i1 < i2);
        bool descSeg = (tid & k) == 0;
        if (descSeg ? !firstBefore : firstBefore) {
          ss[tid] = s2; ss[ixj] = s1;
          si[tid] = i2; si[ixj] = i1;
        }
      }
      __syncthreads();
    }
  }
  candS[tid] = ss[tid]; candI[tid] = si[tid];
}

// ---------------- 7: gather + full decode of the 1024 winners ----------------
__global__ void k_decode(const float* __restrict__ p, const float* __restrict__ ancs,
                         const float* __restrict__ fsz,
                         const float* __restrict__ candS, const int* __restrict__ candI,
                         float* __restrict__ boxK, float* __restrict__ oboxK,
                         float* __restrict__ areaK, int* __restrict__ labelK,
                         int* __restrict__ idK, int* __restrict__ validK) {
  int j = blockIdx.x * blockDim.x + threadIdx.x;
  if (j >= KSLOTS) return;
  int idx = candI[j];
  float sc = candS[j];
  if (idx < 0 || sc <= 0.0f) {
    #pragma unroll
    for (int c = 0; c < 4; ++c) { boxK[j*4+c] = 0.0f; oboxK[j*4+c] = 0.0f; }
    areaK[j] = 0.0f; labelK[j] = -1; idK[j] = -1; validK[j] = 0;
    return;
  }
  int b = idx / NANCH;
  int n = idx - b * NANCH;
  const float* row = p + (long long)idx * 85;
  __builtin_prefetch(row + 32, 0, 0);  // global_prefetch_b8 for the class tail
  float fx = fsz[2*n], fy = fsz[2*n+1];
  float ax = ancs[4*n], ay = ancs[4*n+1], aw = ancs[4*n+2], ah = ancs[4*n+3];
  float cx = ax + sigmoidf_(row[0]) / fx;
  float cy = ay + sigmoidf_(row[1]) / fy;
  float w  = __expf(row[2]) * aw;
  float h  = __expf(row[3]) * ah;
  float l  = cx - 0.5f * w, t  = cy - 0.5f * h;
  float r  = cx + 0.5f * w, bb = cy + 0.5f * h;
  // argmax over class logits (sigmoid is monotonic -> argmax of logits)
  float best = row[5]; int arg = 0;
  for (int c = 1; c < NCLS; ++c) {
    float v = row[5 + c];
    if (v > best) { best = v; arg = c; }
  }
  int label = arg + 1;
  float off = (float)(b * (NCLS + 1) + label) * BOX_OFF;
  boxK[j*4+0] = l;       boxK[j*4+1] = t;       boxK[j*4+2] = r;       boxK[j*4+3] = bb;
  oboxK[j*4+0] = l + off; oboxK[j*4+1] = t + off; oboxK[j*4+2] = r + off; oboxK[j*4+3] = bb + off;
  areaK[j]  = (r - l) * (bb - t);
  labelK[j] = label; idK[j] = b; validK[j] = 1;
}

// ---------------- 8: IoU suppression bitmask, 16x16 tile per wave32 ----------
// Union denominator area_i + area_j computed as a rank-2 WMMA outer product:
//   A(16x4)=[area_row, 1, 0, 0], B(4x16)=[1; area_col; 0; 0] -> D = a_i + a_j
// The threshold test avoids division:  inter/union > thr  <=>  inter > thr*union
// (union = sum - inter + 1e-9 > 0), saving the IEEE divide sequence per pair.
__global__ void __launch_bounds__(32) k_mask(const float* __restrict__ obox,
                                             const float* __restrict__ area,
                                             unsigned short* __restrict__ mask) {
  __shared__ float rb[16][4];
  __shared__ float ra[16];
  int lane = threadIdx.x;
  int i0 = blockIdx.y << 4, j0 = blockIdx.x << 4;
  if (lane < 16) {
    int r = i0 + lane;
    rb[lane][0] = obox[r*4+0];
    rb[lane][1] = obox[r*4+1];
    rb[lane][2] = obox[r*4+2];
    rb[lane][3] = obox[r*4+3];
    ra[lane] = area[r];
  }
  __syncthreads();
  int n = lane & 15;
  float cl = obox[(j0+n)*4+0];
  float ct = obox[(j0+n)*4+1];
  float cr = obox[(j0+n)*4+2];
  float cb = obox[(j0+n)*4+3];
  float ca = area[j0+n];

  v8f d;
#if __has_builtin(__builtin_amdgcn_wmma_f32_16x16x4_f32)
  {
    // f32 A 16x4 layout: lanes 0-15 hold K0(v0),K1(v1); lanes 16-31 hold K2,K3.
    v2f a, bv;
    if (lane < 16) { a = (v2f){ra[lane], 1.0f}; bv = (v2f){1.0f, ca}; }
    else           { a = (v2f){0.0f, 0.0f};     bv = (v2f){0.0f, 0.0f}; }
    v8f c = {};
    d = __builtin_amdgcn_wmma_f32_16x16x4_f32(false, a, false, bv,
                                              (short)0, c, false, false);
  }
#elif __has_builtin(__builtin_amdgcn_wmma_f32_16x16x32_f16)
  {
    v16h a = {}; v16h bv = {};
    if (lane < 16) {
      a[0]  = (_Float16)ra[lane]; a[1]  = (_Float16)1.0f;
      bv[0] = (_Float16)1.0f;     bv[1] = (_Float16)ca;
    }
    v8f c = {};
    d = __builtin_amdgcn_wmma_f32_16x16x32_f16(false, a, false, bv,
                                               (short)0, c, false, false);
  }
#else
  #pragma unroll
  for (int r = 0; r < 8; ++r) {
    int m = r + ((lane & 16) ? 8 : 0);
    d[r] = ra[m] + ca;
  }
#endif

  // C/D layout: vgpr r, lanes 0-15 -> M=r, lanes 16-31 -> M=r+8; N = lane%16.
  #pragma unroll
  for (int r = 0; r < 8; ++r) {
    int m = r + ((lane & 16) ? 8 : 0);
    float il = fmaxf(rb[m][0], cl);
    float it = fmaxf(rb[m][1], ct);
    float ir = fminf(rb[m][2], cr);
    float ib = fminf(rb[m][3], cb);
    float iw = fmaxf(ir - il, 0.0f);
    float ih = fmaxf(ib - it, 0.0f);
    float inter = iw * ih;
    float uni = d[r] - inter + 1e-9f;          // union > 0 always
    bool sup = inter > NMS_THR * uni;          // == (inter/uni > NMS_THR)
#if __has_builtin(__builtin_amdgcn_ballot_w32)
    unsigned bal = __builtin_amdgcn_ballot_w32(sup);
#else
    unsigned bal = (unsigned)__ballot(sup);
#endif
    if (lane == 0) {
      mask[(i0 + r)     * 64 + blockIdx.x] = (unsigned short)(bal & 0xFFFFu);
      mask[(i0 + r + 8) * 64 + blockIdx.x] = (unsigned short)(bal >> 16);
    }
  }
}

// ---------------- 9: sequential NMS + output ----------------
__global__ void __launch_bounds__(KSLOTS) k_nms(const unsigned short* __restrict__ mask,
                                                const int* __restrict__ validK,
                                                const int* __restrict__ idK,
                                                const int* __restrict__ labelK,
                                                const float* __restrict__ boxK,
                                                const float* __restrict__ scoreK,
                                                float* __restrict__ out) {
  __shared__ int keepS[KSLOTS];
  int tid = threadIdx.x;
  keepS[tid] = validK[tid];
  __syncthreads();
  for (int i = 0; i < TOPKN; ++i) {
    if (keepS[i] && tid > i && keepS[tid]) {
      unsigned short w = mask[i * 64 + (tid >> 4)];
      if ((w >> (tid & 15)) & 1) keepS[tid] = 0;
    }
    __syncthreads();
  }
  if (tid < TOPKN) {
    bool kp = keepS[tid] != 0;
    out[tid] = kp ? (float)idK[tid] : -1.0f;                 // ids2
    #pragma unroll
    for (int c = 0; c < 4; ++c)
      out[TOPKN + tid * 4 + c] = kp ? boxK[tid * 4 + c] : 0.0f;  // boxes2
    out[5 * TOPKN + tid] = kp ? (float)labelK[tid] : -1.0f;  // labels2
    out[6 * TOPKN + tid] = kp ? scoreK[tid] : 0.0f;          // scores2
    out[7 * TOPKN + tid] = kp ? 1.0f : 0.0f;                 // keep
  }
}

// ---------------- host launcher ----------------
extern "C" void kernel_launch(void* const* d_in, const int* in_sizes, int n_in,
                              void* d_out, int out_size, void* d_ws, size_t ws_size,
                              hipStream_t stream) {
  (void)in_sizes; (void)n_in; (void)out_size;
  const float* p    = (const float*)d_in[0];   // [32,22743,85]
  const float* ancs = (const float*)d_in[1];   // [22743,4]
  const float* fsz  = (const float*)d_in[2];   // [22743,2]
  float* out = (float*)d_out;                  // 8000 floats
  char* w = (char*)d_ws;

  const long long BN = (long long)BDIM * NANCH;   // 727776
  size_t o = 0;
  float*    scores = (float*)(w + o);    o += (size_t)BN * 4;
  unsigned* hist   = (unsigned*)(w + o); o += 2048 * 4;
  SelState* st     = (SelState*)(w + o); o += 256;
  float*    candS  = (float*)(w + o);    o += KSLOTS * 4;
  int*      candI  = (int*)(w + o);      o += KSLOTS * 4;
  float*    boxK   = (float*)(w + o);    o += KSLOTS * 16;
  float*    oboxK  = (float*)(w + o);    o += KSLOTS * 16;
  float*    areaK  = (float*)(w + o);    o += KSLOTS * 4;
  int*      labelK = (int*)(w + o);      o += KSLOTS * 4;
  int*      idK    = (int*)(w + o);      o += KSLOTS * 4;
  int*      validK = (int*)(w + o);      o += KSLOTS * 4;
  unsigned short* mask = (unsigned short*)(w + o); o += (size_t)KSLOTS * 64 * 2;
  if (o > ws_size) return;  // scratch too small: do nothing (deterministic)

  int blocksBN = (int)((BN + 255) / 256);

  k_zero<<<2, 1024, 0, stream>>>(hist, st);
  k_scores<<<blocksBN, 256, 0, stream>>>(p, scores, BN);
  for (int pass = 0; pass < 3; ++pass) {
    k_hist<<<2048, 256, 0, stream>>>(scores, hist, st, pass, BN);
    k_scan<<<1, 1, 0, stream>>>(hist, st, pass);
  }
  k_initcand<<<1, KSLOTS, 0, stream>>>(candS, candI);
  k_compact<<<blocksBN, 256, 0, stream>>>(scores, candS, candI, st, BN);
  k_sort<<<1, KSLOTS, 0, stream>>>(candS, candI);
  k_decode<<<4, 256, 0, stream>>>(p, ancs, fsz, candS, candI,
                                  boxK, oboxK, areaK, labelK, idK, validK);
  dim3 mg(KSLOTS / 16, KSLOTS / 16);
  k_mask<<<mg, 32, 0, stream>>>(oboxK, areaK, mask);
  k_nms<<<1, KSLOTS, 0, stream>>>(mask, validK, idK, labelK, boxK, candS, out);
}
